// Model_Net_ESM_biLSTM_CombineUpgrade3_23330262351943
// MI455X (gfx1250) — compile-verified, matrix operand order checked by reading, not run on hardware
//
#include <hip/hip_runtime.h>
#include <math.h>

// ---------------------------------------------------------------------------
// MI455X (gfx1250) implementation.
// The dominant GEMM (30000x6464x128, ~50 GFLOP over 778 MB of fp32
// activations) is HBM-bound at 23.3 TB/s (~33us) only if the matmul sustains
// >1.5 PFLOPS -> V_WMMA_F32_16X16X32_BF16 (bf16 operands converted on the fly
// with RNE, f32 accumulate). BK=64: each staged tile feeds 16 WMMAs between
// one barrier pair. Staging packs 2 bf16 per dword (ds_store_b32).
// Edge aggregation uses global_atomic_add_f32. All conv activations are
// padded to K%64==0 and live in the 192MB L2.
// ---------------------------------------------------------------------------

typedef __attribute__((ext_vector_type(16))) __bf16 v16bf;
typedef __attribute__((ext_vector_type(8)))  float  v8f;

__device__ __forceinline__ unsigned short f2bf(float f) {
  union { float f; unsigned int u; } c; c.f = f;
  unsigned int u = c.u;
  unsigned int r = (u + 0x7FFFu + ((u >> 16) & 1u)) >> 16;  // round-nearest-even
  return (unsigned short)r;
}

#define BM 128
#define BN 128
#define BK 64
#define BKD (BK / 2)        // dwords per row (2 bf16 each)
#define LDSROW (BK + 8)     // +8 bf16 pad: 144B row stride, 16B aligned

// C[m, ccol0+n] = (A[m,:Kpad] @ bf16(W[:Kw, n])) (+bias)(+relu)
// A row-major, leading dim lda; caller guarantees A[:, Kw:Kpad] == 0 or
// Kpad == Kw. Kpad must be a multiple of 64.
__global__ __launch_bounds__(256)
void wmma_gemm_kernel(const float* __restrict__ A, int lda, int M,
                      int Kpad, int Kw,
                      const float* __restrict__ W, int Nout,
                      const float* __restrict__ bias,
                      float* __restrict__ C, int ldc, int ccol0, int do_relu)
{
  __shared__ unsigned short sA[BM][LDSROW]  __attribute__((aligned(32)));  // [row][k]
  __shared__ unsigned short sBt[BN][LDSROW] __attribute__((aligned(32)));  // [col][k]

  const int tid   = threadIdx.x;
  const int wave  = tid >> 5;       // 8 waves (wave32)
  const int lane  = tid & 31;
  const int wm    = wave >> 1;      // 0..3  -> 32-row strip
  const int wn    = wave & 1;       // 0..1  -> 64-col strip
  const int laneM = lane & 15;
  const int laneH = lane >> 4;      // K-half selector (contiguous k = 16*laneH + i)

  const int m0 = blockIdx.x * BM;
  const int n0 = blockIdx.y * BN;

  v8f acc[2][4];
  #pragma unroll
  for (int i = 0; i < 2; ++i)
    #pragma unroll
    for (int j = 0; j < 4; ++j)
      #pragma unroll
      for (int r = 0; r < 8; ++r)
        acc[i][j][r] = 0.0f;

  for (int k0 = 0; k0 < Kpad; k0 += BK) {
    // Prefetch next K-tile of A (lowers to global_prefetch_b8 on gfx1250)
    if (k0 + BK < Kpad) {
      int pr = tid >> 1;                     // 0..127 rows
      int pk = (tid & 1) << 5;               // 2 x 128B chunks per row
      int gm = m0 + pr;
      if (gm < M) __builtin_prefetch(&A[(long)gm * lda + (k0 + BK + pk)], 0, 3);
    }

    // Stage A tile (128x64 fp32 -> packed bf16 pairs, ds_store_b32)
    for (int i = tid; i < BM * BKD; i += 256) {
      int r = i >> 5, kd = i & (BKD - 1);    // BKD == 32
      int k = kd << 1;
      int gm = m0 + r;
      float v0 = 0.0f, v1 = 0.0f;
      if (gm < M) {
        const float* ap = &A[(long)gm * lda + (k0 + k)];
        v0 = ap[0]; v1 = ap[1];
      }
      unsigned int pk2 = (unsigned int)f2bf(v0) | ((unsigned int)f2bf(v1) << 16);
      *(unsigned int*)&sA[r][k] = pk2;
    }
    // Stage B tile transposed (64x128 -> LDS [col][k], packed pairs).
    // n fastest across threads -> coalesced global reads along the row.
    for (int i = tid; i < BKD * BN; i += 256) {
      int kd = i >> 7, n = i & 127;
      int k = kd << 1;
      int gk = k0 + k, gn = n0 + n;
      float v0 = 0.0f, v1 = 0.0f;
      if (gn < Nout) {
        if (gk < Kw)     v0 = W[(long)gk * Nout + gn];
        if (gk + 1 < Kw) v1 = W[(long)(gk + 1) * Nout + gn];
      }
      unsigned int pk2 = (unsigned int)f2bf(v0) | ((unsigned int)f2bf(v1) << 16);
      *(unsigned int*)&sBt[n][k] = pk2;
    }
    __syncthreads();

    // Two WMMA K-steps per staged tile. Fragments use contiguous k, same for A and B:
    // A 16x32 bf16 (lanes 0-15 K=0..15, 16-31 K=16..31); B 32x16 (lane = col).
    #pragma unroll
    for (int s = 0; s < BK; s += 32) {
      v16bf af[2], bfr[4];
      #pragma unroll
      for (int i = 0; i < 2; ++i) {
        int row = wm * 32 + i * 16 + laneM;
        af[i] = *(const v16bf*)&sA[row][s + laneH * 16];
      }
      #pragma unroll
      for (int j = 0; j < 4; ++j) {
        int col = wn * 64 + j * 16 + laneM;
        bfr[j] = *(const v16bf*)&sBt[col][s + laneH * 16];
      }
      #pragma unroll
      for (int i = 0; i < 2; ++i)
        #pragma unroll
        for (int j = 0; j < 4; ++j)
          acc[i][j] = __builtin_amdgcn_wmma_f32_16x16x32_bf16(
              false, af[i], false, bfr[j], (short)0, acc[i][j], false, false);
    }
    __syncthreads();
  }

  // C/D layout: lane&15 = N, (lane>>4)*8 + vgpr = M
  #pragma unroll
  for (int i = 0; i < 2; ++i) {
    #pragma unroll
    for (int j = 0; j < 4; ++j) {
      int gn = n0 + wn * 64 + j * 16 + laneM;
      int rowBase = m0 + wm * 32 + i * 16 + laneH * 8;
      #pragma unroll
      for (int r = 0; r < 8; ++r) {
        int gm = rowBase + r;
        if (gm < M && gn < Nout) {
          float v = acc[i][j][r];
          if (bias) v += bias[gn];
          if (do_relu) v = fmaxf(v, 0.0f);
          C[(long)gm * ldc + ccol0 + gn] = v;
        }
      }
    }
  }
}

// ---------------------------------------------------------------------------
// Small elementwise / graph kernels
// ---------------------------------------------------------------------------

__global__ void fill_kernel(float* __restrict__ p, int n, float v) {
  int i = blockIdx.x * blockDim.x + threadIdx.x;
  if (i < n) p[i] = v;
}

// f2 = relu(prot_x[:, :21] @ W_f2 + b); also zero pad cols 149..191 of feat
__global__ void feat_f2_kernel(const float* __restrict__ px,
                               const float* __restrict__ Wf2,
                               const float* __restrict__ bf2,
                               float* __restrict__ feat, int N) {
  int idx = blockIdx.x * blockDim.x + threadIdx.x;
  if (idx >= N * 64) return;
  int i = idx >> 6, jj = idx & 63;
  if (jj < 21) {
    float s = bf2[jj];
    #pragma unroll
    for (int k = 0; k < 21; ++k) s += px[(long)i * 6485 + k] * Wf2[k * 21 + jj];
    feat[(long)i * 192 + jj] = fmaxf(s, 0.0f);
  } else {
    feat[(long)i * 192 + 128 + jj] = 0.0f;   // cols 149..191
  }
}

__global__ void deg_count_kernel(const int* __restrict__ ei, int E,
                                 float* __restrict__ deg) {
  int e = blockIdx.x * blockDim.x + threadIdx.x;
  if (e < E) atomicAdd(&deg[ei[E + e]], 1.0f);   // dst row
}

__global__ void rsqrt_kernel(float* __restrict__ d, int n) {
  int i = blockIdx.x * blockDim.x + threadIdx.x;
  if (i < n) d[i] = rsqrtf(d[i]);
}

// one block per edge: agg[dst,:] += h[src,:] * dinv[src]*dinv[dst]
__global__ __launch_bounds__(128)
void agg_edges_kernel(const int* __restrict__ ei, int E,
                      const float* __restrict__ h, int ldh, int Nout,
                      const float* __restrict__ dinv,
                      float* __restrict__ agg) {
  int e = blockIdx.x;
  if (e >= E) return;
  int src = ei[e], dst = ei[E + e];
  float coef = dinv[src] * dinv[dst];
  const float* hs = h + (long)src * ldh;
  float* ad = agg + (long)dst * ldh;
  for (int c = threadIdx.x; c < Nout; c += 128)
    atomicAdd(&ad[c], hs[c] * coef);
}

// out = relu(agg + h*dinv^2 + b); zero pad columns
__global__ void conv_finalize_kernel(const float* __restrict__ agg,
                                     const float* __restrict__ h,
                                     const float* __restrict__ dinv,
                                     const float* __restrict__ bias,
                                     float* __restrict__ out,
                                     int N, int Nout, int ldc) {
  int idx = blockIdx.x * blockDim.x + threadIdx.x;
  if (idx >= N * ldc) return;
  int i = idx / ldc, j = idx - i * ldc;
  float v = 0.0f;
  if (j < Nout) {
    float di = dinv[i];
    v = fmaxf(agg[idx] + h[idx] * di * di + bias[j], 0.0f);
  }
  out[idx] = v;
}

__global__ void concat_kernel(const float* __restrict__ x2,
                              const float* __restrict__ y2,
                              float* __restrict__ zc, int N) {
  int idx = blockIdx.x * blockDim.x + threadIdx.x;
  if (idx >= N * 640) return;
  int i = idx / 640, j = idx - i * 640;
  float v = 0.0f;
  if (j < 298)      v = x2[(long)i * 320 + j];
  else if (j < 596) v = y2[(long)i * 320 + (j - 298)];
  zc[idx] = v;
}

__global__ void gap_cnt_kernel(const int* __restrict__ batch, int N,
                               float* __restrict__ cnt) {
  int i = blockIdx.x * blockDim.x + threadIdx.x;
  if (i < N) atomicAdd(&cnt[batch[i]], 1.0f);
}

__global__ void gap_sum_kernel(const float* __restrict__ z,
                               const int* __restrict__ batch,
                               float* __restrict__ pooled, int N) {
  int idx = blockIdx.x * blockDim.x + threadIdx.x;
  if (idx >= N * 596) return;
  int i = idx / 596, j = idx - i * 596;
  atomicAdd(&pooled[(long)batch[i] * 640 + j], z[(long)i * 640 + j]);
}

__global__ void gap_div_kernel(float* __restrict__ pooled,
                               const float* __restrict__ cnt, int B) {
  int idx = blockIdx.x * blockDim.x + threadIdx.x;
  if (idx >= B * 596) return;
  int b = idx / 596, j = idx - b * 596;
  pooled[(long)b * 640 + j] /= fmaxf(cnt[b], 1.0f);
}

__global__ void g1_kernel(const float* __restrict__ pooled,
                          const float* __restrict__ Wg1,
                          const float* __restrict__ bg1,
                          float* __restrict__ g1out, int B) {
  int idx = blockIdx.x * blockDim.x + threadIdx.x;
  if (idx >= B * 1024) return;
  int b = idx >> 10, n = idx & 1023;
  float s = bg1[n];
  const float* p = pooled + (long)b * 640;
  for (int k = 0; k < 596; ++k) s += p[k] * Wg1[(long)k * 1024 + n];
  g1out[idx] = s;
}

__global__ void bn_relu_kernel(float* __restrict__ g,
                               const float* __restrict__ gamma,
                               const float* __restrict__ beta, int B) {
  int n = blockIdx.x * blockDim.x + threadIdx.x;
  if (n >= 1024) return;
  float m = 0.0f;
  for (int b = 0; b < 32; ++b) m += g[b * 1024 + n];
  m *= (1.0f / 32.0f);
  float var = 0.0f;
  for (int b = 0; b < 32; ++b) { float d = g[b * 1024 + n] - m; var += d * d; }
  var *= (1.0f / 32.0f);
  float s = rsqrtf(var + 1e-5f) * gamma[n];
  for (int b = 0; b < 32; ++b)
    g[b * 1024 + n] = fmaxf((g[b * 1024 + n] - m) * s + beta[n], 0.0f);
}

__global__ void g2_sigmoid_kernel(const float* __restrict__ g,
                                  const float* __restrict__ Wg2,
                                  const float* __restrict__ bg2,
                                  float* __restrict__ out, int B) {
  int idx = blockIdx.x * blockDim.x + threadIdx.x;
  if (idx >= B * 486) return;
  int b = idx / 486, n = idx - b * 486;
  float s = bg2[n];
  const float* gr = g + (long)b * 1024;
  for (int k = 0; k < 1024; ++k) s += gr[k] * Wg2[(long)k * 486 + n];
  out[idx] = 1.0f / (1.0f + expf(-s));
}

// ---------------------------------------------------------------------------
// Launcher
// ---------------------------------------------------------------------------
extern "C" void kernel_launch(void* const* d_in, const int* in_sizes, int n_in,
                              void* d_out, int out_size, void* d_ws, size_t ws_size,
                              hipStream_t stream) {
  (void)in_sizes; (void)n_in; (void)out_size; (void)ws_size;
  const int N = 30000, E = 300000, B = 32;
  const int NF = 6485, AA = 21, HID = 128;
  const int D1 = 149, D2 = 298, D4 = 596;
  const int LD1 = 192, LD2 = 320, LD4 = 640;   // K-pads (multiples of 64)

  const float* prot_x = (const float*)d_in[0];
  const int*   eiP    = (const int*)d_in[1];
  const int*   eiA    = (const int*)d_in[2];
  const int*   batch  = (const int*)d_in[3];
  const float* W_f1 = (const float*)d_in[4];  const float* b_f1 = (const float*)d_in[5];
  const float* W_f2 = (const float*)d_in[6];  const float* b_f2 = (const float*)d_in[7];
  const float* W_p1 = (const float*)d_in[8];  const float* b_p1 = (const float*)d_in[9];
  const float* W_p2 = (const float*)d_in[10]; const float* b_p2 = (const float*)d_in[11];
  const float* W_a1 = (const float*)d_in[12]; const float* b_a1 = (const float*)d_in[13];
  const float* W_a2 = (const float*)d_in[14]; const float* b_a2 = (const float*)d_in[15];
  const float* W_p3 = (const float*)d_in[16]; const float* b_p3 = (const float*)d_in[17];
  const float* W_g1 = (const float*)d_in[18]; const float* b_g1 = (const float*)d_in[19];
  const float* W_g2 = (const float*)d_in[20]; const float* b_g2 = (const float*)d_in[21];
  const float* gamma = (const float*)d_in[22]; const float* beta = (const float*)d_in[23];

  // workspace carve-out (~355 MB)
  float* ws = (float*)d_ws;
  size_t off = 0;
  auto alloc = [&](size_t n) { float* p = ws + off; off += n; return p; };
  float* feat   = alloc((size_t)N * LD1);
  float* x1     = alloc((size_t)N * LD1);   // also reused as y1
  float* x2     = alloc((size_t)N * LD2);
  float* y2     = alloc((size_t)N * LD2);
  float* zc     = alloc((size_t)N * LD4);
  float* bufH   = alloc((size_t)N * LD4);
  float* bufAgg = alloc((size_t)N * LD4);
  float* dinvP  = alloc(N);
  float* dinvA  = alloc(N);
  float* pooled = alloc((size_t)B * LD4);
  float* cnt    = alloc(B);
  float* g1out  = alloc((size_t)B * 1024);

  dim3 blk(256);
  auto cdiv = [](int a, int b) { return (a + b - 1) / b; };

  // ---- features: f1 (big WMMA GEMM, K=6464 % 64 == 0) + f2 (tiny) ----
  wmma_gemm_kernel<<<dim3(cdiv(N, BM), cdiv(HID, BN)), blk, 0, stream>>>(
      prot_x + AA, NF, N, NF - AA, NF - AA, W_f1, HID, b_f1, feat, LD1, AA, 1);
  feat_f2_kernel<<<cdiv(N * 64, 256), blk, 0, stream>>>(prot_x, W_f2, b_f2, feat, N);

  // ---- symmetric-norm degree terms (deg includes +1 self loop) ----
  fill_kernel<<<cdiv(N, 256), blk, 0, stream>>>(dinvP, N, 1.0f);
  deg_count_kernel<<<cdiv(E, 256), blk, 0, stream>>>(eiP, E, dinvP);
  rsqrt_kernel<<<cdiv(N, 256), blk, 0, stream>>>(dinvP, N);
  fill_kernel<<<cdiv(N, 256), blk, 0, stream>>>(dinvA, N, 1.0f);
  deg_count_kernel<<<cdiv(E, 256), blk, 0, stream>>>(eiA, E, dinvA);
  rsqrt_kernel<<<cdiv(N, 256), blk, 0, stream>>>(dinvA, N);

  // ---- GCN conv: h = in@W ; agg = scatter(h) ; out = relu(agg + h/deg + b) ----
  auto conv = [&](const float* in, int Kpad, int Kw, const float* Wm,
                  const float* bm, int Nout, int ldo, const int* ei,
                  const float* dinv, float* out) {
    wmma_gemm_kernel<<<dim3(cdiv(N, BM), cdiv(Nout, BN)), blk, 0, stream>>>(
        in, Kpad, N, Kpad, Kw, Wm, Nout, nullptr, bufH, ldo, 0, 0);
    fill_kernel<<<cdiv(N * ldo, 256), blk, 0, stream>>>(bufAgg, N * ldo, 0.0f);
    agg_edges_kernel<<<E, dim3(128), 0, stream>>>(ei, E, bufH, ldo, Nout, dinv, bufAgg);
    conv_finalize_kernel<<<cdiv(N * ldo, 256), blk, 0, stream>>>(
        bufAgg, bufH, dinv, bm, out, N, Nout, ldo);
  };

  conv(feat, LD1, D1, W_p1, b_p1, D1, LD1, eiP, dinvP, x1);
  conv(x1,   LD1, D1, W_p2, b_p2, D2, LD2, eiP, dinvP, x2);
  conv(feat, LD1, D1, W_a1, b_a1, D1, LD1, eiA, dinvA, x1);   // y1 reuses x1
  conv(x1,   LD1, D1, W_a2, b_a2, D2, LD2, eiA, dinvA, y2);

  concat_kernel<<<cdiv(N * LD4, 256), blk, 0, stream>>>(x2, y2, zc, N);
  conv(zc, LD4, D4, W_p3, b_p3, D4, LD4, eiP, dinvP, zc);     // in-place safe

  // ---- global average pool + MLP head ----
  fill_kernel<<<cdiv(B * LD4, 256), blk, 0, stream>>>(pooled, B * LD4, 0.0f);
  fill_kernel<<<1, blk, 0, stream>>>(cnt, B, 0.0f);
  gap_cnt_kernel<<<cdiv(N, 256), blk, 0, stream>>>(batch, N, cnt);
  gap_sum_kernel<<<cdiv(N * D4, 256), blk, 0, stream>>>(zc, batch, pooled, N);
  gap_div_kernel<<<cdiv(B * D4, 256), blk, 0, stream>>>(pooled, cnt, B);

  g1_kernel<<<cdiv(B * 1024, 256), blk, 0, stream>>>(pooled, W_g1, b_g1, g1out, B);
  bn_relu_kernel<<<cdiv(1024, 256), blk, 0, stream>>>(g1out, gamma, beta, B);
  g2_sigmoid_kernel<<<cdiv(B * 486, 256), blk, 0, stream>>>(
      g1out, W_g2, b_g2, (float*)d_out, B);
}
